// HGT_57509612093808
// MI455X (gfx1250) — compile-verified, matrix operand order checked by reading, not run on hardware
//
#include <hip/hip_runtime.h>
#include <hip/hip_bf16.h>
#include <math.h>

// ---------------------------------------------------------------------------
// HGT forward for MI455X (gfx1250, wave32). Dense GEMMs via
// v_wmma_f32_16x16x32_bf16 (fp32 accumulate, bf16 staged in LDS, B stored
// transposed so A and B fragments are both contiguous 16B LDS loads);
// edge phase via gathers + hardware fp32 global atomics.
// ---------------------------------------------------------------------------

#define NS_   100000
#define NV_   10000
#define HID_  256
#define H_    8
#define D_    32
#define L_    2

typedef __attribute__((ext_vector_type(16))) __bf16 v16bf;
typedef __attribute__((ext_vector_type(8)))  __bf16 v8bf;
typedef __attribute__((ext_vector_type(8)))  float  v8f;

// ---------------- helpers ----------------
__device__ __forceinline__ float gelu_exact(float x) {
    return 0.5f * x * (1.0f + erff(x * 0.70710678118654752f));
}
__device__ __forceinline__ unsigned enc_ord(float x) {
    unsigned b = __float_as_uint(x);
    return (b & 0x80000000u) ? ~b : (b | 0x80000000u);
}
__device__ __forceinline__ float dec_ord(unsigned u) {
    return (u & 0x80000000u) ? __uint_as_float(u ^ 0x80000000u)
                             : __uint_as_float(~u);
}

// Unified fragment loader (CDNA5 WMMA 16-bit A/B layout, ISA 7.12.2, wave32).
// For lane: row = rowbase + (lane&15); the 16 elements are two contiguous
// 8-element chunks at k = (lane>>4)*8 and k = 16 + (lane>>4)*8.
// Works for A from sA[m][k] and for B from sBt[n][k] (B stored transposed).
__device__ __forceinline__ v16bf load_frag(const __bf16* base, int stride,
                                           int rowbase, int lane) {
    const __bf16* row = base + (size_t)(rowbase + (lane & 15)) * stride;
    const int koff = (lane >> 4) * 8;
    v8bf lo = *(const v8bf*)(row + koff);        // 16B aligned
    v8bf hi = *(const v8bf*)(row + 16 + koff);   // 16B aligned
    v16bf f;
#pragma unroll
    for (int i = 0; i < 8; ++i) { f[i] = lo[i]; f[i + 8] = hi[i]; }
    return f;
}

// ---------------------------------------------------------------------------
// Generic fused GEMM:  C[M,N] = epilogue( act_in(A[M,K]) @ W[K,N] + bias )
//   act_in  : 1 -> exact GELU applied to A while staging
//   act_out : 1 -> ReLU
//   gate    : if non-null, g = sigmoid(*gate); C = g*val + (1-g)*skipx
// Requires N % 64 == 0, K % 32 == 0 (true for all call sites: N=256).
// Block = 128 threads (4 waves); tile 64x64; BK=32.
// ---------------------------------------------------------------------------
__global__ __launch_bounds__(128)
void wmma_gemm_kernel(const float* __restrict__ A, const float* __restrict__ W,
                      const float* __restrict__ bias,
                      const float* __restrict__ skipx,
                      const float* __restrict__ gate,
                      float* __restrict__ C,
                      int M, int Kdim, int N, int act_in, int act_out) {
    __shared__ __bf16 sA[64][40];    // [m][k]   rows 80B-aligned
    __shared__ __bf16 sBt[64][40];   // [n][k]   B transposed

    const int tid  = threadIdx.x;
    const int lane = tid & 31;
    const int wid  = tid >> 5;            // 0..3
    const int wrow = (wid >> 1) * 32;     // wave row offset in tile
    const int wcol = (wid & 1) * 32;      // wave col offset in tile
    const int bm   = blockIdx.x * 64;
    const int bn   = blockIdx.y * 64;

    v8f acc[2][2] = {};

    for (int kb = 0; kb < Kdim; kb += 32) {
        // stage A (64 rows x 32 k), float4 global reads, fp32 -> bf16
        for (int idx = tid; idx < 512; idx += 128) {
            int r = idx >> 3, c4 = (idx & 7) << 2;
            int gr = bm + r;
            float4 v = make_float4(0.f, 0.f, 0.f, 0.f);
            if (gr < M) v = *(const float4*)(A + (size_t)gr * Kdim + kb + c4);
            if (act_in) {
                v.x = gelu_exact(v.x); v.y = gelu_exact(v.y);
                v.z = gelu_exact(v.z); v.w = gelu_exact(v.w);
            }
            __bf16* d = &sA[r][c4];
            d[0] = (__bf16)v.x; d[1] = (__bf16)v.y;
            d[2] = (__bf16)v.z; d[3] = (__bf16)v.w;
        }
        // stage B transposed (32 k-rows x 64 n-cols -> sBt[n][k])
        for (int idx = tid; idx < 512; idx += 128) {
            int r = idx >> 4, c4 = (idx & 15) << 2;
            float4 v = *(const float4*)(W + (size_t)(kb + r) * N + bn + c4);
            sBt[c4 + 0][r] = (__bf16)v.x;
            sBt[c4 + 1][r] = (__bf16)v.y;
            sBt[c4 + 2][r] = (__bf16)v.z;
            sBt[c4 + 3][r] = (__bf16)v.w;
        }
        __syncthreads();

        v16bf afrag[2], bfrag[2];
#pragma unroll
        for (int mi = 0; mi < 2; ++mi)
            afrag[mi] = load_frag(&sA[0][0], 40, wrow + mi * 16, lane);
#pragma unroll
        for (int ni = 0; ni < 2; ++ni)
            bfrag[ni] = load_frag(&sBt[0][0], 40, wcol + ni * 16, lane);
#pragma unroll
        for (int mi = 0; mi < 2; ++mi)
#pragma unroll
            for (int ni = 0; ni < 2; ++ni)
                acc[mi][ni] = __builtin_amdgcn_wmma_f32_16x16x32_bf16(
                    false, afrag[mi], false, bfrag[ni],
                    (short)0, acc[mi][ni], false, false);
        __syncthreads();
    }

    // epilogue. C/D layout: row_local = r + 8*(lane>>4), col_local = lane&15
    float g = 1.0f, gi = 0.0f;
    if (gate) { g = 1.0f / (1.0f + expf(-gate[0])); gi = 1.0f - g; }
#pragma unroll
    for (int mi = 0; mi < 2; ++mi) {
#pragma unroll
        for (int ni = 0; ni < 2; ++ni) {
#pragma unroll
            for (int r = 0; r < 8; ++r) {
                int row = bm + wrow + mi * 16 + r + 8 * (lane >> 4);
                int col = bn + wcol + ni * 16 + (lane & 15);
                if (row < M) {
                    float v = acc[mi][ni][r];
                    if (bias) v += bias[col];
                    if (act_out) v = fmaxf(v, 0.0f);
                    if (gate)
                        v = g * v + gi * skipx[(size_t)row * N + col];
                    C[(size_t)row * N + col] = v;
                }
            }
        }
    }
}

// ---------------------------------------------------------------------------
// Per-head relation transform: Y[n, h*32+e] = sum_d X[n, h*32+d] * Rel[h,d,e]
// grid = (ceil(Nrows/64), H). Block 128 = 4 waves; wave -> 16 rows, K=32=1 WMMA.
// ---------------------------------------------------------------------------
__global__ __launch_bounds__(128)
void head_xform_kernel(const float* __restrict__ X, const float* __restrict__ Rel,
                       float* __restrict__ Y, int Nrows) {
    __shared__ __bf16 sA[64][40];    // [m][d]
    __shared__ __bf16 sWt[32][40];   // [e][d]  relation transposed

    const int tid  = threadIdx.x;
    const int lane = tid & 31;
    const int wid  = tid >> 5;
    const int h    = blockIdx.y;
    const int bm   = blockIdx.x * 64;
    const float* rel = Rel + (size_t)h * D_ * D_;

    // stage relation matrix transposed: sWt[e][d] = rel[d*32+e]
    for (int idx = tid; idx < 256; idx += 128) {
        int r = idx >> 3, c4 = (idx & 7) << 2;          // r = d row, c4 = e col
        float4 v = *(const float4*)(rel + (size_t)r * 32 + c4);
        sWt[c4 + 0][r] = (__bf16)v.x;
        sWt[c4 + 1][r] = (__bf16)v.y;
        sWt[c4 + 2][r] = (__bf16)v.z;
        sWt[c4 + 3][r] = (__bf16)v.w;
    }
    // stage activations (64 rows x 32 d)
    for (int idx = tid; idx < 512; idx += 128) {
        int r = idx >> 3, c4 = (idx & 7) << 2;
        int gr = bm + r;
        float4 v = make_float4(0.f, 0.f, 0.f, 0.f);
        if (gr < Nrows)
            v = *(const float4*)(X + (size_t)gr * HID_ + h * 32 + c4);
        __bf16* d = &sA[r][c4];
        d[0] = (__bf16)v.x; d[1] = (__bf16)v.y;
        d[2] = (__bf16)v.z; d[3] = (__bf16)v.w;
    }
    __syncthreads();

    v16bf afrag = load_frag(&sA[0][0], 40, wid * 16, lane);
    v16bf bfrag[2];
#pragma unroll
    for (int ni = 0; ni < 2; ++ni)
        bfrag[ni] = load_frag(&sWt[0][0], 40, ni * 16, lane);

    v8f acc[2] = {};
#pragma unroll
    for (int ni = 0; ni < 2; ++ni)
        acc[ni] = __builtin_amdgcn_wmma_f32_16x16x32_bf16(
            false, afrag, false, bfrag[ni], (short)0, acc[ni], false, false);

#pragma unroll
    for (int ni = 0; ni < 2; ++ni) {
#pragma unroll
        for (int r = 0; r < 8; ++r) {
            int row = bm + wid * 16 + r + 8 * (lane >> 4);
            int col = h * 32 + ni * 16 + (lane & 15);
            if (row < Nrows) Y[(size_t)row * HID_ + col] = acc[ni][r];
        }
    }
}

// ---------------------------------------------------------------------------
// Edge kernels
// ---------------------------------------------------------------------------
__global__ __launch_bounds__(256)
void fill_u32_kernel(unsigned* __restrict__ buf, unsigned val, int n) {
    int i = blockIdx.x * 256 + threadIdx.x;
    if (i < n) buf[i] = val;
}

// alpha[e*H+h] = dot(Q[dst[e],h,:], KT[src[e],h,:]) * p_rel[h] * scale
__global__ __launch_bounds__(256)
void edge_alpha_kernel(const float* __restrict__ Q, const float* __restrict__ KT,
                       const int* __restrict__ src, const int* __restrict__ dst,
                       const float* __restrict__ prel, float* __restrict__ alpha,
                       int Ecnt) {
    int idx = blockIdx.x * 256 + threadIdx.x;
    if (idx >= Ecnt * H_) return;
    int e = idx >> 3, h = idx & 7;
    const float4* qp = (const float4*)(Q  + (size_t)dst[e] * HID_ + h * 32);
    const float4* kp = (const float4*)(KT + (size_t)src[e] * HID_ + h * 32);
    float acc = 0.0f;
#pragma unroll
    for (int j = 0; j < 8; ++j) {
        float4 q4 = qp[j], k4 = kp[j];
        acc += q4.x * k4.x + q4.y * k4.y + q4.z * k4.z + q4.w * k4.w;
    }
    alpha[idx] = acc * prel[h] * 0.17677669529663687f;   // 1/sqrt(32)
}

__global__ __launch_bounds__(256)
void edge_segmax_kernel(const float* __restrict__ alpha,
                        const int* __restrict__ dst,
                        unsigned* __restrict__ maxb, int Ecnt) {
    int idx = blockIdx.x * 256 + threadIdx.x;
    if (idx >= Ecnt * H_) return;
    int e = idx >> 3, h = idx & 7;
    atomicMax(&maxb[(size_t)dst[e] * H_ + h], enc_ord(alpha[idx]));
}

// ex = exp(alpha - m); den[dst,h] += ex; num[dst,h,:] += ex * VT[src,h,:]
__global__ __launch_bounds__(256)
void edge_accum_kernel(const float* __restrict__ alpha,
                       const float* __restrict__ VT,
                       const int* __restrict__ src, const int* __restrict__ dst,
                       const unsigned* __restrict__ maxb,
                       float* __restrict__ den, float* __restrict__ num,
                       int Ecnt) {
    int idx = blockIdx.x * 256 + threadIdx.x;
    if (idx >= Ecnt * H_) return;
    int e = idx >> 3, h = idx & 7;
    int d = dst[e], s = src[e];
    float m = dec_ord(maxb[(size_t)d * H_ + h]);
    if (!(m > -1e30f)) m = 0.0f;
    float ex = expf(alpha[idx] - m);
    unsafeAtomicAdd(&den[(size_t)d * H_ + h], ex);
    const float4* vp = (const float4*)(VT + (size_t)s * HID_ + h * 32);
    float* np = num + (size_t)d * HID_ + h * 32;
#pragma unroll
    for (int j = 0; j < 8; ++j) {
        float4 v4 = vp[j];
        unsafeAtomicAdd(&np[j * 4 + 0], ex * v4.x);
        unsafeAtomicAdd(&np[j * 4 + 1], ex * v4.y);
        unsafeAtomicAdd(&np[j * 4 + 2], ex * v4.z);
        unsafeAtomicAdd(&np[j * 4 + 3], ex * v4.w);
    }
}

__global__ __launch_bounds__(256)
void finalize_agg_kernel(const float* __restrict__ num,
                         const float* __restrict__ den,
                         float* __restrict__ agg, int Nd) {
    int idx = blockIdx.x * 256 + threadIdx.x;
    if (idx >= Nd * HID_) return;
    int n = idx >> 8, h = (idx & 255) >> 5;
    agg[idx] += num[idx] / (den[(size_t)n * H_ + h] + 1e-16f);
}

// ---------------------------------------------------------------------------
// Scorer
// ---------------------------------------------------------------------------
// bias2[j] = b1[j] + sum_k veh_last[k] * W1[256+k, j]
__global__ __launch_bounds__(256)
void scorer_bias2_kernel(const float* __restrict__ veh_last,
                         const float* __restrict__ w1,
                         const float* __restrict__ b1,
                         float* __restrict__ bias2) {
    __shared__ float sv[256];
    int j = threadIdx.x;
    sv[j] = veh_last[j];
    __syncthreads();
    float acc = b1[j];
    for (int k = 0; k < 256; ++k)
        acc += sv[k] * w1[(size_t)(256 + k) * 256 + j];
    bias2[j] = acc;
}

__global__ __launch_bounds__(256)
void scorer_final_kernel(const float* __restrict__ Hb,
                         const float* __restrict__ w2,
                         const float* __restrict__ b2,
                         float* __restrict__ out, int M) {
    __shared__ float sw[512];
    int tid = threadIdx.x;
    sw[tid] = w2[tid];
    sw[tid + 256] = w2[tid + 256];
    __syncthreads();
    int n = blockIdx.x * 256 + tid;
    if (n >= M) return;
    const float* hr = Hb + (size_t)n * 256;
    float a0 = 0.0f, a1 = 0.0f;
    for (int c = 0; c < 256; ++c) {
        float h = hr[c];
        a0 += h * sw[2 * c];
        a1 += h * sw[2 * c + 1];
    }
    out[n]     = a0 + b2[0];
    out[M + n] = 1.0f / (1.0f + expf(-(a1 + b2[1])));
}

// ---------------------------------------------------------------------------
// Host orchestration
// ---------------------------------------------------------------------------
extern "C" void kernel_launch(void* const* d_in, const int* in_sizes, int n_in,
                              void* d_out, int out_size, void* d_ws, size_t ws_size,
                              hipStream_t stream) {
    const float* x_stop  = (const float*)d_in[0];
    const float* x_veh   = (const float*)d_in[1];
    const int*   e_ss    = (const int*)d_in[2];
    const int*   e_vs    = (const int*)d_in[3];
    const int*   e_sv    = (const int*)d_in[4];
    const float* lin_w   = (const float*)d_in[5];
    const float* lin_b   = (const float*)d_in[6];
    const float* kqv_w   = (const float*)d_in[7];
    const float* kqv_b   = (const float*)d_in[8];
    const float* alin_w  = (const float*)d_in[9];
    const float* alin_b  = (const float*)d_in[10];
    const float* skip    = (const float*)d_in[11];
    const float* a_rel   = (const float*)d_in[12];
    const float* m_rel   = (const float*)d_in[13];
    const float* p_rel   = (const float*)d_in[14];
    const float* w1      = (const float*)d_in[15];
    const float* b1      = (const float*)d_in[16];
    const float* w2      = (const float*)d_in[17];
    const float* b2      = (const float*)d_in[18];
    float* out = (float*)d_out;

    // ---- workspace layout (floats); total ~221M floats (~884 MB) ----
    const size_t nsF = (size_t)NS_ * HID_;   // 25,600,000
    const size_t nvF = (size_t)NV_ * HID_;   //  2,560,000
    float* ws  = (float*)d_ws;
    float* XS0 = ws;                 float* XS1 = XS0 + nsF;
    float* K0  = XS1 + nvF;          float* Q0  = K0  + nsF;
    float* V0  = Q0  + nsF;          float* K1  = V0  + nsF;
    float* Q1  = K1  + nvF;          float* V1  = Q1  + nvF;
    float* KT  = V1  + nvF;          float* VT  = KT  + nsF;
    float* NUMB = VT + nsF;          float* AGG0 = NUMB + nsF;
    float* AGG1 = AGG0 + nsF;
    float* ALPHA = AGG1 + nvF;                       // 1,600,000
    float* MAXB  = ALPHA + 1600000;                  //   800,000 (as u32)
    float* DENB  = MAXB + 800000;                    //   800,000
    float* BIAS2 = DENB + 800000;                    //       256
    float* SH    = Q0;  // scorer hidden reuses Q0 after layers

    const int NNt[2]  = {NS_, NV_};
    const int SRCT[3] = {0, 1, 0};
    const int DSTT[3] = {0, 0, 1};
    const int ECNT[3] = {200000, 100000, 100000};
    const int* eptr[3] = {e_ss, e_vs, e_sv};
    float* XSb[2]  = {XS0, XS1};
    float* Kb[2]   = {K0, K1};
    float* Qb[2]   = {Q0, Q1};
    float* Vb[2]   = {V0, V1};
    float* AGGb[2] = {AGG0, AGG1};

    dim3 blk128(128), blk256(256);
    auto gemm_grid = [](int M) { return dim3((M + 63) / 64, HID_ / 64); };

    // ---- stage 0: per-type input projection + ReLU (K=64) ----
    for (int t = 0; t < 2; ++t) {
        hipLaunchKernelGGL(wmma_gemm_kernel, gemm_grid(NNt[t]), blk128, 0, stream,
                           t ? x_veh : x_stop, lin_w + (size_t)t * 64 * HID_,
                           lin_b + (size_t)t * HID_, nullptr, nullptr,
                           XSb[t], NNt[t], 64, HID_, 0, 1);
    }

    // ---- HGT layers ----
    for (int l = 0; l < L_; ++l) {
        // K/Q/V projections: [N,256] x [256,256] + bias
        for (int p = 0; p < 3; ++p) {
            float* dstb[2];
            dstb[0] = (p == 0) ? K0 : (p == 1) ? Q0 : V0;
            dstb[1] = (p == 0) ? K1 : (p == 1) ? Q1 : V1;
            for (int t = 0; t < 2; ++t) {
                size_t wi = ((size_t)(l * 3 + p) * 2 + t);
                hipLaunchKernelGGL(wmma_gemm_kernel, gemm_grid(NNt[t]), blk128, 0, stream,
                                   XSb[t], kqv_w + wi * HID_ * HID_,
                                   kqv_b + wi * HID_, nullptr, nullptr,
                                   dstb[t], NNt[t], HID_, HID_, 0, 0);
            }
        }

        hipMemsetAsync(AGG0, 0, nsF * sizeof(float), stream);
        hipMemsetAsync(AGG1, 0, nvF * sizeof(float), stream);

        for (int et = 0; et < 3; ++et) {
            const int s = SRCT[et], d = DSTT[et];
            const int Nsrc = NNt[s], Nd = NNt[d], Ecnt = ECNT[et];
            const int* srcp = eptr[et];
            const int* dstp = eptr[et] + Ecnt;
            const size_t reloff = (size_t)(l * 3 + et) * H_ * D_ * D_;

            // relation-transformed keys/values (per-head 32x32 WMMA)
            hipLaunchKernelGGL(head_xform_kernel, dim3((Nsrc + 63) / 64, H_),
                               blk128, 0, stream, Kb[s], a_rel + reloff, KT, Nsrc);
            hipLaunchKernelGGL(head_xform_kernel, dim3((Nsrc + 63) / 64, H_),
                               blk128, 0, stream, Vb[s], m_rel + reloff, VT, Nsrc);

            int nEH = Ecnt * H_;
            hipLaunchKernelGGL(edge_alpha_kernel, dim3((nEH + 255) / 256),
                               blk256, 0, stream, Qb[d], KT, srcp, dstp,
                               p_rel + (size_t)(l * 3 + et) * H_, ALPHA, Ecnt);

            int nDH = Nd * H_;
            hipLaunchKernelGGL(fill_u32_kernel, dim3((nDH + 255) / 256),
                               blk256, 0, stream, (unsigned*)MAXB,
                               0x007FFFFFu /* enc(-inf) */, nDH);
            hipLaunchKernelGGL(edge_segmax_kernel, dim3((nEH + 255) / 256),
                               blk256, 0, stream, ALPHA, dstp, (unsigned*)MAXB, Ecnt);

            hipMemsetAsync(DENB, 0, (size_t)nDH * sizeof(float), stream);
            hipMemsetAsync(NUMB, 0, (size_t)Nd * HID_ * sizeof(float), stream);
            hipLaunchKernelGGL(edge_accum_kernel, dim3((nEH + 255) / 256),
                               blk256, 0, stream, ALPHA, VT, srcp, dstp,
                               (const unsigned*)MAXB, DENB, NUMB, Ecnt);

            int nDC = Nd * HID_;
            hipLaunchKernelGGL(finalize_agg_kernel, dim3((nDC + 255) / 256),
                               blk256, 0, stream, NUMB, DENB, AGGb[d], Nd);
        }

        // o = gelu(agg) @ a_lin_w + b;  xs = g*o + (1-g)*xs   (in place)
        for (int t = 0; t < 2; ++t) {
            size_t wi = (size_t)(l * 2 + t);
            hipLaunchKernelGGL(wmma_gemm_kernel, gemm_grid(NNt[t]), blk128, 0, stream,
                               AGGb[t], alin_w + wi * HID_ * HID_,
                               alin_b + wi * HID_, XSb[t], skip + wi,
                               XSb[t], NNt[t], HID_, HID_, 1, 0);
        }
    }

    // ---- scorer ----
    // fold veh_last @ W1[256:512] + b1 into a single bias vector
    hipLaunchKernelGGL(scorer_bias2_kernel, dim3(1), blk256, 0, stream,
                       XS1 + (size_t)(NV_ - 1) * HID_, w1, b1, BIAS2);
    // h = relu(xs0 @ W1[0:256] + bias2)  (top half of w1 is contiguous)
    hipLaunchKernelGGL(wmma_gemm_kernel, gemm_grid(NS_), blk128, 0, stream,
                       XS0, w1, BIAS2, nullptr, nullptr, SH, NS_, HID_, HID_, 0, 1);
    // o = h @ W2 + b2 ; out = [o0, sigmoid(o1)]
    hipLaunchKernelGGL(scorer_final_kernel, dim3((NS_ + 255) / 256),
                       blk256, 0, stream, SH, w2, b2, out, NS_);
}
